// VectorQuantizer_71442486001893
// MI455X (gfx1250) — matrix operand order repaired and checked by reading, not compiled, not run on hardware
//
#include <hip/hip_runtime.h>
#include <hip/hip_bf16.h>
#include <math.h>

// ---------------------------------------------------------------------------
// VectorQuantizer for MI455X (gfx1250):
//  - v_wmma_f32_16x16x32_bf16 for the two big matmuls (dist + ortho)
//  - global_load_async_to_lds_b128 (ASYNCcnt) double-buffered codebook tiles
//  - 32 rows/wave M-blocking: 1 ds_load_b128 per WMMA (LDS-bandwidth balanced)
// ---------------------------------------------------------------------------

typedef __attribute__((ext_vector_type(16))) __bf16       v16bf;
typedef __attribute__((ext_vector_type(8)))  float        v8f;
typedef __attribute__((ext_vector_type(8)))  unsigned int v8u;
typedef __attribute__((ext_vector_type(4)))  unsigned int v4u;

#define VQ_NE     8192
#define VQ_C      256
#define VQ_NROWS  16384      // 16*32*32
#define VQ_HW     1024       // 32*32
#define VQ_BETA   0.25f
#define VQ_NCHUNK 4          // codebook split for the dist kernel
#define VQ_TPC    (VQ_NE / 16 / VQ_NCHUNK)   // 128 entry-tiles per chunk
#define VQ_RPB    256        // z rows per block in vq_dist (32 per wave)

// round-to-nearest-even f32 -> bf16 (bit pattern)
__device__ __forceinline__ unsigned short f2bf(float f) {
  unsigned int u = __float_as_uint(f);
  u += 0x7fffu + ((u >> 16) & 1u);
  return (unsigned short)(u >> 16);
}
__device__ __forceinline__ unsigned int pack2bf(float lo, float hi) {
  return (unsigned int)f2bf(lo) | ((unsigned int)f2bf(hi) << 16);
}

// B-fragment (32x16 bf16, K x N): lane = N | (khalf<<4); 16 contiguous K per lane.
__device__ __forceinline__ v16bf load_bfrag(const unsigned short* p) {
  return __builtin_bit_cast(v16bf, *(const v8u*)p);
}
// A-fragment (16x32 bf16, M x K): lane = M | (khalf<<4); 16 contiguous K per lane.
__device__ __forceinline__ v16bf load_afrag(const unsigned short* p) {
  return __builtin_bit_cast(v16bf, *(const v8u*)p);
}

// ---- CDNA5 async global->LDS copy (inline asm; tracked by ASYNCcnt) --------
__device__ __forceinline__ void async_g2l_b128(unsigned int lds_byte_addr,
                                               unsigned long long gaddr) {
  asm volatile("global_load_async_to_lds_b128 %0, %1, off"
               :: "v"(lds_byte_addr), "v"(gaddr) : "memory");
}
__device__ __forceinline__ void wait_async0() {
  asm volatile("s_wait_asynccnt 0" ::: "memory");
}

__device__ __forceinline__ float waveReduceSum(float v) {
  #pragma unroll
  for (int m = 16; m >= 1; m >>= 1) v += __shfl_xor(v, m, 32);
  return v;
}

__device__ __forceinline__ float blockReduceSum(float v) {
  __shared__ float red[8];
  v = waveReduceSum(v);
  int wave = threadIdx.x >> 5, lane = threadIdx.x & 31;
  if (lane == 0) red[wave] = v;
  __syncthreads();
  if (wave == 0) {
    v = (lane < 8) ? red[lane] : 0.0f;
    #pragma unroll
    for (int m = 4; m >= 1; m >>= 1) v += __shfl_xor(v, m, 32);
  }
  return v;  // valid on thread 0
}

// ---------------------------------------------------------------------------
// 1) prep: per codebook row: ||e||^2, bf16 codebook, bf16 normalized codebook
// ---------------------------------------------------------------------------
__global__ void __launch_bounds__(256) vq_prep(const float* __restrict__ cb,
                                               unsigned short* __restrict__ cb_bf,
                                               unsigned short* __restrict__ wn_bf,
                                               float* __restrict__ cnorm) {
  int wave = threadIdx.x >> 5, lane = threadIdx.x & 31;
  int row  = blockIdx.x * 8 + wave;
  const float* src = cb + (size_t)row * VQ_C + lane * 8;
  float4 a = *(const float4*)src;
  float4 b = *(const float4*)(src + 4);
  float ss = a.x*a.x + a.y*a.y + a.z*a.z + a.w*a.w
           + b.x*b.x + b.y*b.y + b.z*b.z + b.w*b.w;
  ss = waveReduceSum(ss);
  if (lane == 0) cnorm[row] = ss;
  float inv = rsqrtf(ss);

  v4u pc, pw;
  pc.x = pack2bf(a.x, a.y);         pc.y = pack2bf(a.z, a.w);
  pc.z = pack2bf(b.x, b.y);         pc.w = pack2bf(b.z, b.w);
  pw.x = pack2bf(a.x*inv, a.y*inv); pw.y = pack2bf(a.z*inv, a.w*inv);
  pw.z = pack2bf(b.x*inv, b.y*inv); pw.w = pack2bf(b.z*inv, b.w*inv);
  *(v4u*)(cb_bf + (size_t)row * VQ_C + lane * 8) = pc;
  *(v4u*)(wn_bf + (size_t)row * VQ_C + lane * 8) = pw;
}

// ---------------------------------------------------------------------------
// 2) distance + partial argmin.
//    Block = 256 z rows (wave w owns rows +w*32: two 16-row A tiles resident)
//    x one codebook chunk (2048 entries). B tiles (16 entries x 256 bf16 =
//    8KB) staged to LDS with async loads, double-buffered, shared by all 8
//    waves, and each B fragment feeds two WMMAs (1 ds_load_b128 per WMMA).
// ---------------------------------------------------------------------------
__global__ void __launch_bounds__(256) vq_dist(const float* __restrict__ z,
                                               const unsigned short* __restrict__ cb_bf,
                                               const float* __restrict__ cnorm,
                                               float* __restrict__ partial_v,
                                               int* __restrict__ partial_i) {
  __shared__ __align__(32) unsigned short zt_bf[VQ_RPB * VQ_C];  // 128 KB
  __shared__ __align__(32) unsigned short bb[2][16 * VQ_C];      // 2 x 8 KB

  int tid   = threadIdx.x;
  int rblk  = blockIdx.x;             // 0..63 : 256 z rows each
  int chunk = blockIdx.y;             // 0..3  : 2048 codebook entries each
  int tbase = chunk * VQ_TPC;
  int wave  = tid >> 5, lane = tid & 31;

  // stage 256 z rows as bf16 into LDS (coalesced over hw)
  for (int i = tid; i < VQ_RPB * VQ_C; i += 256) {
    int m = i & (VQ_RPB - 1), c = i >> 8;
    int p = rblk * VQ_RPB + m;        // global flat row (b*1024 + hw)
    float v = z[(size_t)(p >> 10) * (VQ_C * VQ_HW) + (size_t)c * VQ_HW + (p & 1023)];
    zt_bf[m * VQ_C + c] = f2bf(v);
  }

  int half   = lane >> 4;
  int mrow   = (lane & 15);
  int nloc   = (lane & 15);
  int kbaseA = half * 16;
  int koffB  = half * 16;

  // prologue: every wave async-copies 2 x 512B of tile tbase into buffer 0
  {
    #pragma unroll
    for (int i = 0; i < 2; ++i) {
      int ch = wave * 2 + i;          // 16 x 512B chunks per 8KB tile
      unsigned int       la = (unsigned int)(size_t)(&bb[0][0] + ch * 256 + lane * 8);
      unsigned long long ga = (unsigned long long)(size_t)
          (cb_bf + (size_t)tbase * 4096 + ch * 256 + lane * 8);
      async_g2l_b128(la, ga);
    }
  }
  wait_async0();
  __syncthreads();                    // z staged + tile 0 staged

  // per-wave A fragments: rows wave*32 + mrow (set A) and +16 (set B)
  v16bf afA[8], afB[8];
  #pragma unroll
  for (int s = 0; s < 8; ++s) {
    int kk = s * 32;
    afA[s] = load_afrag(&zt_bf[(wave * 32 + mrow)      * VQ_C + kk + kbaseA]);
    afB[s] = load_afrag(&zt_bf[(wave * 32 + 16 + mrow) * VQ_C + kk + kbaseA]);
  }

  float bvA[8], bvB[8];
  int   biA[8], biB[8];
  #pragma unroll
  for (int r = 0; r < 8; ++r) {
    bvA[r] = 3.4e38f; biA[r] = 0;
    bvB[r] = 3.4e38f; biB[r] = 0;
  }

  for (int t = tbase; t < tbase + VQ_TPC; ++t) {
    int cur = (t - tbase) & 1;
    // prefetch next tile into the other buffer (last read in the previous iteration)
    if (t + 1 < tbase + VQ_TPC) {
      #pragma unroll
      for (int i = 0; i < 2; ++i) {
        int ch = wave * 2 + i;
        unsigned int       la = (unsigned int)(size_t)(&bb[cur ^ 1][0] + ch * 256 + lane * 8);
        unsigned long long ga = (unsigned long long)(size_t)
            (cb_bf + (size_t)(t + 1) * 4096 + ch * 256 + lane * 8);
        async_g2l_b128(la, ga);
      }
    }

    // compute on current tile: each B fragment reused by two WMMAs,
    // four independent accumulation chains
    const unsigned short* brow = &bb[cur][nloc * VQ_C + koffB];
    v8f aA0 = {}, aA1 = {}, aB0 = {}, aB1 = {};
    #pragma unroll
    for (int s = 0; s < 4; ++s) {
      v16bf b0 = load_bfrag(brow + (2 * s) * 32);
      v16bf b1 = load_bfrag(brow + (2 * s + 1) * 32);
      aA0 = __builtin_amdgcn_wmma_f32_16x16x32_bf16(false, afA[2 * s],     false, b0,
                                                    (short)0, aA0, false, false);
      aB0 = __builtin_amdgcn_wmma_f32_16x16x32_bf16(false, afB[2 * s],     false, b0,
                                                    (short)0, aB0, false, false);
      aA1 = __builtin_amdgcn_wmma_f32_16x16x32_bf16(false, afA[2 * s + 1], false, b1,
                                                    (short)0, aA1, false, false);
      aB1 = __builtin_amdgcn_wmma_f32_16x16x32_bf16(false, afB[2 * s + 1], false, b1,
                                                    (short)0, aB1, false, false);
    }

    float cn = cnorm[t * 16 + nloc];
    int   ng = t * 16 + nloc;
    #pragma unroll
    for (int r = 0; r < 8; ++r) {
      float dA = cn - 2.0f * (aA0[r] + aA1[r]);   // ||z||^2 dropped (const/row)
      float dB = cn - 2.0f * (aB0[r] + aB1[r]);
      if (dA < bvA[r]) { bvA[r] = dA; biA[r] = ng; }
      if (dB < bvB[r]) { bvB[r] = dB; biB[r] = ng; }
    }

    wait_async0();                    // own async issues for t+1 complete
    __syncthreads();                  // all staged; all done reading bb[cur]
  }

  // argmin across the 16 lanes of each half (xor masks stay within half)
  #pragma unroll
  for (int r = 0; r < 8; ++r) {
    float vA = bvA[r]; int iA = biA[r];
    float vB = bvB[r]; int iB = biB[r];
    #pragma unroll
    for (int m = 8; m >= 1; m >>= 1) {
      float ovA = __shfl_xor(vA, m, 32);
      int   oiA = __shfl_xor(iA, m, 32);
      if (ovA < vA || (ovA == vA && oiA < iA)) { vA = ovA; iA = oiA; }
      float ovB = __shfl_xor(vB, m, 32);
      int   oiB = __shfl_xor(iB, m, 32);
      if (ovB < vB || (ovB == vB && oiB < iB)) { vB = ovB; iB = oiB; }
    }
    bvA[r] = vA; biA[r] = iA;
    bvB[r] = vB; biB[r] = iB;
  }
  if ((lane & 15) == 0) {             // lanes 0 and 16: rows m = r + 8*half
    #pragma unroll
    for (int r = 0; r < 8; ++r) {
      int rowA = rblk * VQ_RPB + wave * 32 + 8 * half + r;
      partial_v[chunk * VQ_NROWS + rowA] = bvA[r];
      partial_i[chunk * VQ_NROWS + rowA] = biA[r];
      int rowB = rowA + 16;
      partial_v[chunk * VQ_NROWS + rowB] = bvB[r];
      partial_i[chunk * VQ_NROWS + rowB] = biB[r];
    }
  }
}

// ---------------------------------------------------------------------------
// 2b) combine the per-chunk partial argmins; build histogram
// ---------------------------------------------------------------------------
__global__ void __launch_bounds__(256) vq_argmin(const float* __restrict__ pv,
                                                 const int* __restrict__ pi,
                                                 int* __restrict__ idx_out,
                                                 unsigned int* __restrict__ hist) {
  int row = blockIdx.x * 256 + threadIdx.x;
  float v = pv[row]; int ix = pi[row];
  #pragma unroll
  for (int c = 1; c < VQ_NCHUNK; ++c) {
    float ov = pv[c * VQ_NROWS + row];
    int   oi = pi[c * VQ_NROWS + row];
    if (ov < v) { v = ov; ix = oi; }  // strict <: ties keep lowest index
  }
  idx_out[row] = ix;
  atomicAdd(&hist[ix], 1u);
}

// ---------------------------------------------------------------------------
// 3) gather z_q, write transposed output, accumulate sum((z_q - z)^2)
// ---------------------------------------------------------------------------
__global__ void __launch_bounds__(256) vq_gather(const float* __restrict__ z,
                                                 const float* __restrict__ cb,
                                                 const int* __restrict__ idx,
                                                 float* __restrict__ out,
                                                 float* __restrict__ accum) {
  int b = blockIdx.x, c = blockIdx.y, tid = threadIdx.x;
  float local = 0.0f;
  for (int hw = tid; hw < VQ_HW; hw += 256) {
    int id = idx[b * VQ_HW + hw];
    float zq = cb[(size_t)id * VQ_C + c];
    size_t off = (size_t)b * (VQ_C * VQ_HW) + (size_t)c * VQ_HW + hw;
    float d = zq - z[off];
    out[off] = zq;                     // z_q_st == z_q forward
    local += d * d;
  }
  local = blockReduceSum(local);
  if (tid == 0) atomicAdd(&accum[0], local);
}

// ---------------------------------------------------------------------------
// 4) ortho: sum((wn wn^T - I)^2) via WMMA; symmetry: only tj >= ti, x2 weight
// ---------------------------------------------------------------------------
__global__ void __launch_bounds__(256) vq_ortho(const unsigned short* __restrict__ wn_bf,
                                                float* __restrict__ accum) {
  int wave = threadIdx.x >> 5, lane = threadIdx.x & 31;
  int ti = blockIdx.x;                // 0..511
  int tj = blockIdx.y * 8 + wave;     // 0..511
  if (tj < ti) return;                // wave-uniform early exit (symmetry)

  int half = lane >> 4;
  int mrow = lane & 15, nloc = lane & 15;
  const unsigned short* arow = wn_bf + ((size_t)(ti * 16 + mrow)) * VQ_C + half * 16;
  const unsigned short* brow = wn_bf + ((size_t)(tj * 16 + nloc)) * VQ_C + half * 16;

  v8f acc0 = {}, acc1 = {};
  #pragma unroll
  for (int s = 0; s < 4; ++s) {
    v16bf a0 = load_afrag(arow + (2 * s) * 32);
    v16bf b0 = load_bfrag(brow + (2 * s) * 32);
    v16bf a1 = load_afrag(arow + (2 * s + 1) * 32);
    v16bf b1 = load_bfrag(brow + (2 * s + 1) * 32);
    acc0 = __builtin_amdgcn_wmma_f32_16x16x32_bf16(false, a0, false, b0,
                                                   (short)0, acc0, false, false);
    acc1 = __builtin_amdgcn_wmma_f32_16x16x32_bf16(false, a1, false, b1,
                                                   (short)0, acc1, false, false);
  }

  float s = 0.0f;
  int ng = tj * 16 + nloc;
  #pragma unroll
  for (int r = 0; r < 8; ++r) {
    int mg = ti * 16 + 8 * half + r;
    float g = (acc0[r] + acc1[r]) - (mg == ng ? 1.0f : 0.0f);
    s += g * g;
  }
  s = waveReduceSum(s);
  float w = (ti == tj) ? 1.0f : 2.0f;  // off-diagonal tiles counted twice
  if (lane == 0) atomicAdd(&accum[1], w * s);
}

// ---------------------------------------------------------------------------
// 5) finalize scalars: loss, ortho_reg_term, perplexity
// ---------------------------------------------------------------------------
__global__ void __launch_bounds__(256) vq_finalize(const unsigned int* __restrict__ hist,
                                                   const float* __restrict__ accum,
                                                   float* __restrict__ out_scalars) {
  int tid = threadIdx.x;
  float s = 0.0f;
  for (int e = tid; e < VQ_NE; e += 256) {
    float em = (float)hist[e] / (float)VQ_NROWS;
    s += em * logf(em + 1e-10f);
  }
  s = blockReduceSum(s);
  if (tid == 0) {
    out_scalars[0] = (1.0f + VQ_BETA) * accum[0] / (float)(VQ_NROWS * VQ_C);
    out_scalars[1] = accum[1] / ((float)VQ_NE * (float)VQ_NE);
    out_scalars[2] = expf(-s);
  }
}

__global__ void __launch_bounds__(256) vq_idx_out(const int* __restrict__ idx,
                                                  float* __restrict__ out) {
  int i = blockIdx.x * 256 + threadIdx.x;
  out[i] = (float)idx[i];
}

// ---------------------------------------------------------------------------
extern "C" void kernel_launch(void* const* d_in, const int* in_sizes, int n_in,
                              void* d_out, int out_size, void* d_ws, size_t ws_size,
                              hipStream_t stream) {
  const float* z  = (const float*)d_in[0];   // (16,256,32,32) f32
  const float* cb = (const float*)d_in[1];   // (8192,256) f32
  float* out = (float*)d_out;

  // workspace layout (bytes)
  char* ws = (char*)d_ws;
  unsigned short* cb_bf = (unsigned short*)(ws);                 // 4 MB
  unsigned short* wn_bf = (unsigned short*)(ws + 4194304);       // 4 MB
  float*          cnorm = (float*)(ws + 8388608);                // 32 KB
  int*            idxb  = (int*)(ws + 8421376);                  // 64 KB
  unsigned int*   hist  = (unsigned int*)(ws + 8486912);         // 32 KB
  float*          accum = (float*)(ws + 8519680);                // 8 B
  float*          pv    = (float*)(ws + 8519936);                // 256 KB
  int*            pi    = (int*)(ws + 8782080);                  // 256 KB

  // zero histogram + the two scalar accumulators (contiguous)
  hipMemsetAsync(hist, 0, VQ_NE * sizeof(unsigned int) + 2 * sizeof(float), stream);

  vq_prep<<<VQ_NE / 8, 256, 0, stream>>>(cb, cb_bf, wn_bf, cnorm);
  vq_dist<<<dim3(VQ_NROWS / VQ_RPB, VQ_NCHUNK), 256, 0, stream>>>(z, cb_bf, cnorm, pv, pi);
  vq_argmin<<<VQ_NROWS / 256, 256, 0, stream>>>(pv, pi, idxb, hist);
  vq_gather<<<dim3(16, VQ_C), 256, 0, stream>>>(z, cb, idxb, out, accum);
  vq_ortho<<<dim3(VQ_NE / 16, VQ_NE / 16 / 8), 256, 0, stream>>>(wn_bf, accum);
  vq_finalize<<<1, 256, 0, stream>>>(hist, accum, out + 4194304);
  vq_idx_out<<<VQ_NROWS / 256, 256, 0, stream>>>(idxb, out + 4194307);
}
